// PolarOut_89764816486664
// MI455X (gfx1250) — matrix operand order repaired and checked by reading, not compiled, not run on hardware
//
#include <hip/hip_runtime.h>
#include <math.h>

typedef __attribute__((ext_vector_type(16))) _Float16 v16h;
typedef __attribute__((ext_vector_type(8)))  float    v8f;

#define WPB 4            // waves per block (wave32)
#define TPB (WPB * 32)

__device__ __forceinline__ float fsigmoid(float x) {
    return 1.0f / (1.0f + __expf(-x));
}

struct SharedStore {
    alignas(32) _Float16 Wt_s1[64 * 128];   // transposed [n][k], f16
    alignas(32) _Float16 Wt_e0[64 * 128];   // transposed [n][k], f16
    float W_g[64 * 16];
    float W_e2[32 * 16];
    float W_s2[64 * 2];
    float W_o0[64];
    float W_o2[16];
    float b_s1[64];
    float b_e0[64];
    float b_g[16];
    float b_s2[2];
    float b_o0;
    float hbuf[WPB][16 * 65];               // padded stride 65: conflict-free column reads
};

// 16x64 = (16 rows, K=128) x (128,64) GEMM via 16 x v_wmma_f32_16x16x32_f16.
// arow: per-lane pointer to row (this lane's M = lane&15), f32 in global.
// Wt:   f16 weights transposed [n][k] in LDS.
__device__ __forceinline__ void gemm_tile_k128(const float* __restrict__ arow,
                                               const _Float16* __restrict__ Wt,
                                               int g, int mloc, v8f acc[4]) {
#pragma unroll
    for (int kk = 0; kk < 4; ++kk) {
        // A fragment, contiguous k order (same as B below):
        // lane group g: element e -> K = 32kk + 16g + e
        const float* ap = arow + kk * 32 + g * 16;
        const float4 a0 = *(const float4*)(ap + 0);
        const float4 a1 = *(const float4*)(ap + 4);
        const float4 a2 = *(const float4*)(ap + 8);
        const float4 a3 = *(const float4*)(ap + 12);
        v16h a;
        a[0]  = (_Float16)a0.x; a[1]  = (_Float16)a0.y; a[2]  = (_Float16)a0.z; a[3]  = (_Float16)a0.w;
        a[4]  = (_Float16)a1.x; a[5]  = (_Float16)a1.y; a[6]  = (_Float16)a1.z; a[7]  = (_Float16)a1.w;
        a[8]  = (_Float16)a2.x; a[9]  = (_Float16)a2.y; a[10] = (_Float16)a2.z; a[11] = (_Float16)a2.w;
        a[12] = (_Float16)a3.x; a[13] = (_Float16)a3.y; a[14] = (_Float16)a3.z; a[15] = (_Float16)a3.w;
#pragma unroll
        for (int c = 0; c < 4; ++c) {
            // B fragment 32x16: lane N = mloc + 16c, element e -> K = 32kk + 16g + e
            const v16h b = *(const v16h*)(Wt + (c * 16 + mloc) * 128 + kk * 32 + g * 16);
            acc[c] = __builtin_amdgcn_wmma_f32_16x16x32_f16(
                false, a, false, b, (short)0, acc[c], false, false);
        }
    }
}

__global__ __launch_bounds__(TPB) void polar_zero_kernel(float* __restrict__ p, int n) {
    const int i = blockIdx.x * blockDim.x + threadIdx.x;
    if (i < n) p[i] = 0.0f;
}

__global__ __launch_bounds__(TPB) void polar_node_kernel(
    const float* __restrict__ node_scalar, const float* __restrict__ node_equi,
    const int* __restrict__ batch,
    const float* __restrict__ gW_s1, const float* __restrict__ gb_s1,
    const float* __restrict__ gW_s2, const float* __restrict__ gb_s2,
    const float* __restrict__ gW_e0, const float* __restrict__ gb_e0,
    const float* __restrict__ gW_e2, const float* __restrict__ gW_g,
    const float* __restrict__ gb_g,  const float* __restrict__ gW_o0,
    const float* __restrict__ gb_o0, const float* __restrict__ gW_o2,
    float* __restrict__ polar, int Nn, int numTiles)
{
    __shared__ SharedStore sm;
    const int tid = threadIdx.x;

    // ---- cooperative weight staging (f16-transposed big weights, f32 smalls) ----
    for (int i = tid; i < 128 * 64; i += TPB) {
        const int k = i >> 6, n = i & 63;          // source row-major [k][n]
        sm.Wt_s1[n * 128 + k] = (_Float16)gW_s1[i];
        sm.Wt_e0[n * 128 + k] = (_Float16)gW_e0[i];
    }
    for (int i = tid; i < 64 * 16; i += TPB) sm.W_g[i]  = gW_g[i];
    for (int i = tid; i < 32 * 16; i += TPB) sm.W_e2[i] = gW_e2[i];
    for (int i = tid; i < 64 * 2;  i += TPB) sm.W_s2[i] = gW_s2[i];
    if (tid < 64) { sm.W_o0[tid] = gW_o0[tid]; sm.b_s1[tid] = gb_s1[tid]; sm.b_e0[tid] = gb_e0[tid]; }
    if (tid < 16) { sm.W_o2[tid] = gW_o2[tid]; sm.b_g[tid] = gb_g[tid]; }
    if (tid < 2)  sm.b_s2[tid] = gb_s2[tid];
    if (tid == 0) sm.b_o0 = gb_o0[0];
    __syncthreads();

    const int wave = tid >> 5;
    const int lane = tid & 31;
    const int g    = lane >> 4;     // half-wave group
    const int mloc = lane & 15;     // local row / column index
    const int tileIdx = blockIdx.x * WPB + wave;
    if (tileIdx >= numTiles) return;

    const int row  = tileIdx * 16 + mloc;
    const int rowc = row < Nn ? row : (Nn - 1);
    float* hb = sm.hbuf[wave];      // wave-private scratch, no block barrier needed

    const v8f zero8 = {0.f, 0.f, 0.f, 0.f, 0.f, 0.f, 0.f, 0.f};

    // ================= s-branch: silu(ns @ W_s1 + b) @ W_s2 + b =================
    v8f accs[4];
#pragma unroll
    for (int c = 0; c < 4; ++c) accs[c] = zero8;
    gemm_tile_k128(node_scalar + (size_t)rowc * 128, sm.Wt_s1, g, mloc, accs);

#pragma unroll
    for (int c = 0; c < 4; ++c) {
        const int n = c * 16 + mloc;
#pragma unroll
        for (int r = 0; r < 8; ++r) {
            const int m = r + 8 * g;                 // C layout: lanes 0-15 M=r, 16-31 M=r+8
            const float v = accs[c][r] + sm.b_s1[n];
            hb[m * 65 + n] = v * fsigmoid(v);        // silu
        }
    }
    // lane (mloc, g) computes s[mloc][g]
    float sv = sm.b_s2[g];
    for (int k = 0; k < 64; ++k) sv += hb[mloc * 65 + k] * sm.W_s2[k * 2 + g];
    const float s_other = __shfl_xor(sv, 16);        // lanes<16: s1 ; lanes>=16: s0

    // ================= equi-branch: h0 = x0 @ W_e0 + b_e0 =======================
    v8f acce[4];
#pragma unroll
    for (int c = 0; c < 4; ++c) acce[c] = zero8;
    gemm_tile_k128(node_equi + (size_t)rowc * 480, sm.Wt_e0, g, mloc, acce);

#pragma unroll
    for (int c = 0; c < 4; ++c) {
        const int n = c * 16 + mloc;
#pragma unroll
        for (int r = 0; r < 8; ++r) {
            const int m = r + 8 * g;
            hb[m * 65 + n] = acce[c][r] + sm.b_e0[n];   // pre-activation h0
        }
    }

    // gate = sigmoid(h0 @ W_g + b_g): lane covers h = g*8 .. g*8+7 of its node
    float gatev[8];
#pragma unroll
    for (int i = 0; i < 8; ++i) {
        const int nh = g * 8 + i;
        float a = sm.b_g[nh];
        for (int k = 0; k < 64; ++k) a += hb[mloc * 65 + k] * sm.W_g[k * 16 + nh];
        gatev[i] = fsigmoid(a);
    }

    // e0 = silu(h0) @ W_o0 + b_o0 (halves split over g, combined via shuffle)
    float e0p = 0.0f;
    for (int k = 0; k < 32; ++k) {
        const int n = g * 32 + k;
        const float v = hb[mloc * 65 + n];
        e0p += (v * fsigmoid(v)) * sm.W_o0[n];
    }
    e0p += __shfl_xor(e0p, 16);
    const float e0 = e0p + sm.b_o0;

    // h2 = einsum('mv,mh->hv', x2, W_e2) ; gated ; e2 = einsum('hv,h->v', h2, W_o2)
    float h2r[8][5];
#pragma unroll
    for (int i = 0; i < 8; ++i)
#pragma unroll
        for (int v = 0; v < 5; ++v) h2r[i][v] = 0.0f;

    const float* x2p = node_equi + (size_t)rowc * 480 + 320;
    for (int mm = 0; mm < 32; ++mm) {
        float xv[5];
#pragma unroll
        for (int v = 0; v < 5; ++v) xv[v] = x2p[mm * 5 + v];
#pragma unroll
        for (int i = 0; i < 8; ++i) {
            const float w = sm.W_e2[mm * 16 + g * 8 + i];
#pragma unroll
            for (int v = 0; v < 5; ++v) h2r[i][v] += xv[v] * w;
        }
    }
    float e2p[5] = {0.f, 0.f, 0.f, 0.f, 0.f};
#pragma unroll
    for (int i = 0; i < 8; ++i) {
        const float t = gatev[i] * sm.W_o2[g * 8 + i];
#pragma unroll
        for (int v = 0; v < 5; ++v) e2p[v] += h2r[i][v] * t;
    }
#pragma unroll
    for (int v = 0; v < 5; ++v) e2p[v] += __shfl_xor(e2p[v], 16);

    // ================= segment-sum accumulate (lanes 0..15 own the 16 nodes) ====
    if (lane < 16 && row < Nn) {
        const int gidx = batch[row];
        float* dst = polar + (size_t)gidx * 6;
        atomicAdd(dst + 0, e0 * sv);                 // sv == s[m][0] on lanes<16
        const float s1 = s_other;                    // s[m][1]
#pragma unroll
        for (int v = 0; v < 5; ++v) atomicAdd(dst + 1 + v, e2p[v] * s1);
    }
}

__global__ __launch_bounds__(256) void polar_finalize_kernel(
    const float* __restrict__ polar, float* __restrict__ out, int G)
{
    const int gi = blockIdx.x * blockDim.x + threadIdx.x;
    if (gi >= G) return;
    const float* p = polar + (size_t)gi * 6;
    const float zero = p[0];
    const float dxy = p[1], dyz = p[2], dz2 = p[3], dzx = p[4], dx2y2 = p[5];
    const float dn = sqrtf(dxy * dxy + dyz * dyz + dz2 * dz2 + dzx * dzx + dx2y2 * dx2y2);
    const float c = 0.57735026918962576f;            // 1/sqrt(3)
    const float a00 = zero + c * (dn - dz2) + dx2y2;
    const float a11 = zero + c * (dn - dz2) - dx2y2;
    const float a22 = zero + c * (dn + 2.0f * dz2);
    float* o = out + (size_t)gi * 9;
    o[0] = a00; o[1] = dxy; o[2] = dzx;
    o[3] = dxy; o[4] = a11; o[5] = dyz;
    o[6] = dzx; o[7] = dyz; o[8] = a22;
}

extern "C" void kernel_launch(void* const* d_in, const int* in_sizes, int n_in,
                              void* d_out, int out_size, void* d_ws, size_t ws_size,
                              hipStream_t stream) {
    const float* node_scalar = (const float*)d_in[0];
    const float* node_equi   = (const float*)d_in[1];
    const int*   batch       = (const int*)d_in[2];
    // d_in[3] = n_graphs (device scalar) — G derived host-side from out_size instead
    const float* W_s1 = (const float*)d_in[4];
    const float* b_s1 = (const float*)d_in[5];
    const float* W_s2 = (const float*)d_in[6];
    const float* b_s2 = (const float*)d_in[7];
    const float* W_e0 = (const float*)d_in[8];
    const float* b_e0 = (const float*)d_in[9];
    const float* W_e2 = (const float*)d_in[10];
    const float* W_g  = (const float*)d_in[11];
    const float* b_g  = (const float*)d_in[12];
    const float* W_o0 = (const float*)d_in[13];
    const float* b_o0 = (const float*)d_in[14];
    const float* W_o2 = (const float*)d_in[15];

    const int Nn = in_sizes[0] / 128;   // node count
    const int G  = out_size / 9;        // graph count (output is G x 3 x 3)

    float* polar = (float*)d_ws;        // G*6 f32 accumulator in workspace
    const int accN = G * 6;
    polar_zero_kernel<<<(accN + TPB - 1) / TPB, TPB, 0, stream>>>(polar, accN);

    const int numTiles = (Nn + 15) / 16;
    const int blocks   = (numTiles + WPB - 1) / WPB;
    polar_node_kernel<<<blocks, TPB, 0, stream>>>(
        node_scalar, node_equi, batch,
        W_s1, b_s1, W_s2, b_s2, W_e0, b_e0, W_e2, W_g, b_g, W_o0, b_o0, W_o2,
        polar, Nn, numTiles);

    polar_finalize_kernel<<<(G + 255) / 256, 256, 0, stream>>>(polar, (float*)d_out, G);
}
